// EncoderLayer_54116587929733
// MI455X (gfx1250) — compile-verified, matrix operand order checked by reading, not run on hardware
//
#include <hip/hip_runtime.h>
#include <hip/hip_bf16.h>
#include <stdint.h>

#ifndef __has_builtin
#define __has_builtin(x) 0
#endif

// ---------------------------------------------------------------------------
// CDNA5 async global->LDS copy (ASYNCcnt-tracked), with safe fallback.
// Builtin signature (from compiler diagnostic): first param is a pointer to a
// 16B int-vector in the global (__device__/AS1) address space; LDS dst in AS3.
// ---------------------------------------------------------------------------
typedef int async_i4 __attribute__((vector_size(16)));

#if __has_builtin(__builtin_amdgcn_global_load_async_to_lds_b128)
#define HAVE_ASYNC_LDS 1
#define ASYNC_COPY16(g, l)                                                    \
    __builtin_amdgcn_global_load_async_to_lds_b128(                           \
        (__attribute__((address_space(1))) async_i4*)(g),                     \
        (__attribute__((address_space(3))) async_i4*)(l), 0, 0)
#if __has_builtin(__builtin_amdgcn_s_wait_asynccnt)
#define WAIT_ASYNC(lit) __builtin_amdgcn_s_wait_asynccnt(lit)
#else
#define WAIT_ASYNC(lit) asm volatile("s_wait_asynccnt " #lit ::: "memory")
#endif
#else
#define HAVE_ASYNC_LDS 0
#define ASYNC_COPY16(g, l) (*(uint4*)(l) = *(const uint4*)(g))
#define WAIT_ASYNC(lit) ((void)0)
#endif

// ---------------------------------------------------------------------------
// Types / helpers
// ---------------------------------------------------------------------------
typedef __attribute__((ext_vector_type(16))) __bf16 v16bf;
typedef __attribute__((ext_vector_type(8)))  float  v8f;

union FragU {
    uint4 q[2];   // 32 bytes
    v16bf v;      // 16 x bf16
};

static __device__ __forceinline__ unsigned short f32_to_bf16_bits(float f) {
    union { float f; uint32_t u; } x; x.f = f;
    uint32_t u = x.u;
    uint32_t r = u + 0x7FFFu + ((u >> 16) & 1u);   // round-to-nearest-even
    return (unsigned short)(r >> 16);
}

static __device__ __forceinline__ float wave_sum(float v) {
    #pragma unroll
    for (int o = 16; o > 0; o >>= 1) v += __shfl_xor(v, o, 32);
    return v;
}
static __device__ __forceinline__ float wave_max(float v) {
    #pragma unroll
    for (int o = 16; o > 0; o >>= 1) v = fmaxf(v, __shfl_xor(v, o, 32));
    return v;
}

// ---------------------------------------------------------------------------
// Elementwise convert / transpose
// ---------------------------------------------------------------------------
__global__ void k_f32_to_bf16(const float* __restrict__ src,
                              unsigned short* __restrict__ dst, int n) {
    int i = blockIdx.x * blockDim.x + threadIdx.x;
    if (i < n) dst[i] = f32_to_bf16_bits(src[i]);
}

// dst[e*256 + d] = bf16(src[d*256 + e])   (A^T for the scores GEMM)
__global__ void k_transpose256_bf16(const float* __restrict__ src,
                                    unsigned short* __restrict__ dst) {
    int i = blockIdx.x * blockDim.x + threadIdx.x;
    int d = i & 255, e = i >> 8;
    if (e < 256) dst[(size_t)e * 256 + d] = f32_to_bf16_bits(src[(size_t)d * 256 + e]);
}

// ---------------------------------------------------------------------------
// Batched bf16 WMMA GEMM, double-buffered LDS, async global->LDS staging.
//   A: [M,K] bf16 row-major (lda), batch offset zb*sAb + zh*sAh
//   B: TRANSB==0 -> [K,N] row-major ; TRANSB==1 -> [N,K] row-major (ldb)
//   C: CBF16 ? bf16 : f32, row-major (ldc)
// Block tile 128x128x32; 256 threads = 8 wave32; wave tile 32(M) x 64(N).
// ---------------------------------------------------------------------------
#define BM 128
#define BN 128
#define BK 32
#define LDSP 56   // LDS row pitch (bf16 elems): 112B rows, 16B aligned, conflict-free

struct GemmParams {
    const unsigned short* A; long long lda, sAb, sAh;
    const unsigned short* B; long long ldb, sBb, sBh;
    void* C; long long ldc, sCb, sCh;
    const float* bias;
    int M, N, K, Zh;
    float alpha;
};

template <int TRANSB, int CBF16, int RELU>
__global__ __launch_bounds__(256) void k_gemm_bf16(GemmParams p) {
    __shared__ unsigned short ldsA[2][BM * LDSP];
    __shared__ unsigned short ldsB[2][BN * LDSP];

    const int z  = blockIdx.z;
    const int zb = z / p.Zh, zh = z % p.Zh;
    const unsigned short* A = p.A + (long long)zb * p.sAb + (long long)zh * p.sAh;
    const unsigned short* B = p.B + (long long)zb * p.sBb + (long long)zh * p.sBh;
    const long long offC = (long long)zb * p.sCb + (long long)zh * p.sCh;

    const int tid  = threadIdx.x;
    const int wave = tid >> 5, lane = tid & 31;
    const int wm = wave >> 1, wn = wave & 1;          // 4x2 wave grid
    const int lhalf = lane >> 4, l16 = lane & 15;

    const int m0 = blockIdx.y * BM;
    const int n0 = blockIdx.x * BN;

    // --- staging thread mappings (hoisted, incremental addressing) ---
    const int rowH = tid >> 1, hv = tid & 1;          // A / TRANSB-B: 2 thr/row, 32B each
    const unsigned short* srcA0 = A + (long long)(m0 + rowH) * p.lda + hv * 16;
    const int ldsOffH = rowH * LDSP + hv * 16;

    const unsigned short* srcB0 = nullptr;
    int nB = 0, khB = 0;
    if (TRANSB) {
        srcB0 = B + (long long)(n0 + rowH) * p.ldb + hv * 16;
    } else {
        nB = tid & 127; khB = tid >> 7;               // thread owns col n, half of K
        srcB0 = B + (long long)khB * 16 * p.ldb + (n0 + nB);
    }

    unsigned short breg[16];                          // reg-buffered B (TRANSB==0)

    auto stageA = [&](int buf, int k0) {
        const unsigned short* s = srcA0 + k0;
        unsigned short* d = &ldsA[buf][ldsOffH];
        ASYNC_COPY16(s, d);
        ASYNC_COPY16(s + 8, d + 8);
        if (TRANSB) {
            const unsigned short* sb = srcB0 + k0;
            unsigned short* db = &ldsB[buf][ldsOffH];
            ASYNC_COPY16(sb, db);
            ASYNC_COPY16(sb + 8, db + 8);
        }
    };
    auto loadB = [&](int k0) {                        // TRANSB==0: issue global loads
        if (!TRANSB) {
            const unsigned short* s = srcB0 + (long long)k0 * p.ldb;
            #pragma unroll
            for (int kk = 0; kk < 16; ++kk) { breg[kk] = *s; s += p.ldb; }
        }
    };
    auto storeB = [&](int buf) {                      // TRANSB==0: pack + 2x ds_store_b128
        if (!TRANSB) {
            unsigned int w[8];
            #pragma unroll
            for (int i = 0; i < 8; ++i)
                w[i] = (unsigned int)breg[2 * i] | ((unsigned int)breg[2 * i + 1] << 16);
            unsigned short* d = &ldsB[buf][nB * LDSP + khB * 16];
            uint4 q0; q0.x = w[0]; q0.y = w[1]; q0.z = w[2]; q0.w = w[3];
            uint4 q1; q1.x = w[4]; q1.y = w[5]; q1.z = w[6]; q1.w = w[7];
            *(uint4*)(d)     = q0;
            *(uint4*)(d + 8) = q1;
        }
    };

    v8f acc[2][4];
    {
        v8f vz = {0.f, 0.f, 0.f, 0.f, 0.f, 0.f, 0.f, 0.f};
        #pragma unroll
        for (int i = 0; i < 2; ++i)
            #pragma unroll
            for (int j = 0; j < 4; ++j) acc[i][j] = vz;
    }

    const int nT = p.K / BK;
    // prologue: stage tile 0
    stageA(0, 0);
    loadB(0);
    storeB(0);

    for (int it = 0; it < nT; ++it) {
        const int cur = it & 1;
        const bool more = (it + 1 < nT);
        if (more) {                                   // prefetch tile it+1
            stageA(cur ^ 1, (it + 1) * BK);
            loadB((it + 1) * BK);
        }
#if HAVE_ASYNC_LDS
        // async completes in order: <=ops_of_one_stage outstanding => stage(it) resident
        if (more) { if (TRANSB) WAIT_ASYNC(4); else WAIT_ASYNC(2); }
        else      { WAIT_ASYNC(0); }
#endif
        __syncthreads();                              // stage(it) visible to all waves

        v16bf af[2], bfr[4];
        #pragma unroll
        for (int i = 0; i < 2; ++i) {
            const unsigned short* pA = &ldsA[cur][(wm * 32 + i * 16 + l16) * LDSP + lhalf * 8];
            FragU f; f.q[0] = *(const uint4*)(pA); f.q[1] = *(const uint4*)(pA + 16);
            af[i] = f.v;
        }
        #pragma unroll
        for (int j = 0; j < 4; ++j) {
            const unsigned short* pB = &ldsB[cur][(wn * 64 + j * 16 + l16) * LDSP + lhalf * 8];
            FragU f; f.q[0] = *(const uint4*)(pB); f.q[1] = *(const uint4*)(pB + 16);
            bfr[j] = f.v;
        }
        #pragma unroll
        for (int i = 0; i < 2; ++i)
            #pragma unroll
            for (int j = 0; j < 4; ++j)
                acc[i][j] = __builtin_amdgcn_wmma_f32_16x16x32_bf16(
                    false, af[i], false, bfr[j], (short)0, acc[i][j], false, false);

        if (more) storeB(cur ^ 1);                    // B regs -> next LDS buffer
        __syncthreads();                              // everyone done with buf cur
    }

    // ---- epilogue: C/D layout lane=N, vgpr r -> M = r + 8*(lane>=16) ----
    #pragma unroll
    for (int i = 0; i < 2; ++i) {
        #pragma unroll
        for (int j = 0; j < 4; ++j) {
            int n = n0 + wn * 64 + j * 16 + l16;
            float bv = p.bias ? p.bias[n] : 0.f;
            #pragma unroll
            for (int r = 0; r < 8; ++r) {
                int m = m0 + wm * 32 + i * 16 + r + lhalf * 8;
                float v = acc[i][j][r] * p.alpha + bv;
                if (RELU) v = fmaxf(v, 0.f);
                long long ci = offC + (long long)m * p.ldc + n;
                if (CBF16) ((unsigned short*)p.C)[ci] = f32_to_bf16_bits(v);
                else       ((float*)p.C)[ci] = v;
            }
        }
    }
}

// ---------------------------------------------------------------------------
// Row softmax over S=2048 (in-place f32, plus bf16 copy). 1 block per row.
// ---------------------------------------------------------------------------
__global__ __launch_bounds__(256) void k_softmax_row2048(float* __restrict__ sc,
                                                         unsigned short* __restrict__ ab) {
    __shared__ float red[8];
    const long long base = (long long)blockIdx.x * 2048;
    const int t = threadIdx.x, wave = t >> 5, lane = t & 31;
    float vals[8];
    float lmax = -INFINITY;
    #pragma unroll
    for (int i = 0; i < 8; ++i) { vals[i] = sc[base + t + i * 256]; lmax = fmaxf(lmax, vals[i]); }
    lmax = wave_max(lmax);
    if (lane == 0) red[wave] = lmax;
    __syncthreads();
    float gmax = red[0];
    #pragma unroll
    for (int i = 1; i < 8; ++i) gmax = fmaxf(gmax, red[i]);
    __syncthreads();
    float lsum = 0.f;
    #pragma unroll
    for (int i = 0; i < 8; ++i) { vals[i] = __expf(vals[i] - gmax); lsum += vals[i]; }
    lsum = wave_sum(lsum);
    if (lane == 0) red[wave] = lsum;
    __syncthreads();
    float gsum = 0.f;
    #pragma unroll
    for (int i = 0; i < 8; ++i) gsum += red[i];
    float inv = 1.f / gsum;
    #pragma unroll
    for (int i = 0; i < 8; ++i) {
        float v = vals[i] * inv;
        sc[base + t + i * 256] = v;
        ab[base + t + i * 256] = f32_to_bf16_bits(v);
    }
}

// ---------------------------------------------------------------------------
// LayerNorm (row length 256, block 256 threads, 1 elem/thread)
// ---------------------------------------------------------------------------
static __device__ __forceinline__ float block_mean256(float v, float* red, int tid) {
    int wave = tid >> 5, lane = tid & 31;
    float s = wave_sum(v);
    __syncthreads();
    if (lane == 0) red[wave] = s;
    __syncthreads();
    float tot = 0.f;
    #pragma unroll
    for (int i = 0; i < 8; ++i) tot += red[i];
    return tot * (1.0f / 256.0f);
}

__global__ __launch_bounds__(256) void k_ln_dual(const float* __restrict__ tmp,
                                                 const float* __restrict__ x,
                                                 const float* __restrict__ g1, const float* __restrict__ b1,
                                                 const float* __restrict__ g2, const float* __restrict__ b2,
                                                 float* __restrict__ y2f, unsigned short* __restrict__ y2b) {
    __shared__ float red[8];
    const long long base = (long long)blockIdx.x * 256;
    const int t = threadIdx.x;
    float xv = x[base + t];
    float v = tmp[base + t] + xv;
    float mu = block_mean256(v, red, t);
    float dv = v - mu;
    float var = block_mean256(dv * dv, red, t);
    v = dv * rsqrtf(var + 1e-5f) * g1[t] + b1[t];
    v = v + xv;
    mu = block_mean256(v, red, t);
    dv = v - mu;
    var = block_mean256(dv * dv, red, t);
    v = dv * rsqrtf(var + 1e-5f) * g2[t] + b2[t];
    y2f[base + t] = v;
    y2b[base + t] = f32_to_bf16_bits(v);
}

__global__ __launch_bounds__(256) void k_ln_final(const float* __restrict__ ffn,
                                                  const float* __restrict__ y2,
                                                  const float* __restrict__ g, const float* __restrict__ b,
                                                  float* __restrict__ out) {
    __shared__ float red[8];
    const long long base = (long long)blockIdx.x * 256;
    const int t = threadIdx.x;
    float v = ffn[base + t] + y2[base + t];
    float mu = block_mean256(v, red, t);
    float dv = v - mu;
    float var = block_mean256(dv * dv, red, t);
    out[base + t] = dv * rsqrtf(var + 1e-5f) * g[t] + b[t];
}

// ---------------------------------------------------------------------------
// Host-side orchestration
// ---------------------------------------------------------------------------
static void launch_gemm(hipStream_t stream,
                        const unsigned short* A, long long lda, long long sAb, long long sAh,
                        const unsigned short* B, long long ldb, long long sBb, long long sBh, int transB,
                        void* C, int c_bf16, long long ldc, long long sCb, long long sCh,
                        const float* bias, int relu, float alpha,
                        int M, int N, int K, int Z, int Zh) {
    GemmParams p;
    p.A = A; p.lda = lda; p.sAb = sAb; p.sAh = sAh;
    p.B = B; p.ldb = ldb; p.sBb = sBb; p.sBh = sBh;
    p.C = C; p.ldc = ldc; p.sCb = sCb; p.sCh = sCh;
    p.bias = bias; p.M = M; p.N = N; p.K = K;
    p.Zh = (Zh < 1) ? 1 : Zh; p.alpha = alpha;
    dim3 grid(N / BN, M / BM, Z);
#define GEMM_CASE(TB, CB, RL)                                                  \
    k_gemm_bf16<TB, CB, RL><<<grid, 256, 0, stream>>>(p)
    if (transB) {
        if (c_bf16) { if (relu) GEMM_CASE(1, 1, 1); else GEMM_CASE(1, 1, 0); }
        else        { if (relu) GEMM_CASE(1, 0, 1); else GEMM_CASE(1, 0, 0); }
    } else {
        if (c_bf16) { if (relu) GEMM_CASE(0, 1, 1); else GEMM_CASE(0, 1, 0); }
        else        { if (relu) GEMM_CASE(0, 0, 1); else GEMM_CASE(0, 0, 0); }
    }
#undef GEMM_CASE
}

extern "C" void kernel_launch(void* const* d_in, const int* in_sizes, int n_in,
                              void* d_out, int out_size, void* d_ws, size_t ws_size,
                              hipStream_t stream) {
    (void)in_sizes; (void)n_in; (void)out_size; (void)ws_size;
    const int B = 8, S = 2048, D = 256, H = 8, HID = 2048;
    const long long BS = (long long)B * S;          // 16384 rows

    const float* X     = (const float*)d_in[0];
    const float* Wq    = (const float*)d_in[2];
    const float* Wk    = (const float*)d_in[3];
    const float* Wv    = (const float*)d_in[4];
    const float* Wo    = (const float*)d_in[5];
    const float* Amat  = (const float*)d_in[6];
    const float* gAttn = (const float*)d_in[7];
    const float* bAttn = (const float*)d_in[8];
    const float* gEnc  = (const float*)d_in[9];
    const float* bEnc  = (const float*)d_in[10];
    const float* W1    = (const float*)d_in[11];
    const float* b1    = (const float*)d_in[12];
    const float* W2    = (const float*)d_in[13];
    const float* b2    = (const float*)d_in[14];
    const float* gFfn  = (const float*)d_in[15];
    const float* bFfn  = (const float*)d_in[16];

    float* outP  = (float*)d_out;                       // [B,S,D]
    float* attnF = outP + (long long)B * S * D;         // [B,H,D,S]

    char* wp = (char*)d_ws;
    auto carve = [&](size_t bytes) -> void* {
        void* p = (void*)wp;
        wp += (bytes + 255) & ~(size_t)255;
        return p;
    };
    unsigned short* Xb    = (unsigned short*)carve(BS * D * 2);
    unsigned short* Wqb   = (unsigned short*)carve((size_t)D * HID * 2);
    unsigned short* Wkb   = (unsigned short*)carve((size_t)D * HID * 2);
    unsigned short* Wvb   = (unsigned short*)carve((size_t)D * HID * 2);
    unsigned short* Wob   = (unsigned short*)carve((size_t)HID * D * 2);
    unsigned short* W1b   = (unsigned short*)carve((size_t)D * HID * 2);
    unsigned short* W2b   = (unsigned short*)carve((size_t)HID * D * 2);
    unsigned short* Ab    = (unsigned short*)carve((size_t)D * D * 2);
    unsigned short* AbT   = (unsigned short*)carve((size_t)D * D * 2);
    unsigned short* Qb    = (unsigned short*)carve(BS * HID * 2);
    unsigned short* Kb    = (unsigned short*)carve(BS * HID * 2);            // reused: ctx
    unsigned short* Vb    = (unsigned short*)carve(BS * HID * 2);            // reused: ffn hidden
    unsigned short* attnB = (unsigned short*)carve((size_t)B * H * D * S * 2); // reused: QA
    unsigned short* VVb   = (unsigned short*)carve((size_t)B * H * D * D * 2);
    float* tmpF = (float*)carve(BS * D * 4);
    float* y2F  = (float*)carve(BS * D * 4);
    unsigned short* y2b = (unsigned short*)carve(BS * D * 2);
    float* ffnF = (float*)carve(BS * D * 4);
    unsigned short* ctxB = Kb;
    unsigned short* hB   = Vb;
    unsigned short* QAb  = attnB;

    auto cvt = [&](const float* s, unsigned short* dd, long long n) {
        k_f32_to_bf16<<<(unsigned)((n + 255) / 256), 256, 0, stream>>>(s, dd, (int)n);
    };

    // 1. bf16 conversions
    cvt(X, Xb, BS * D);
    cvt(Wq, Wqb, (long long)D * HID);
    cvt(Wk, Wkb, (long long)D * HID);
    cvt(Wv, Wvb, (long long)D * HID);
    cvt(Wo, Wob, (long long)HID * D);
    cvt(W1, W1b, (long long)D * HID);
    cvt(W2, W2b, (long long)HID * D);
    cvt(Amat, Ab, (long long)D * D);
    k_transpose256_bf16<<<256, 256, 0, stream>>>(Amat, AbT);

    // 2. Q/K/V projections
    launch_gemm(stream, Xb, D, 0, 0, Wqb, HID, 0, 0, 0, Qb, 1, HID, 0, 0,
                nullptr, 0, 1.f, (int)BS, HID, D, 1, 1);
    launch_gemm(stream, Xb, D, 0, 0, Wkb, HID, 0, 0, 0, Kb, 1, HID, 0, 0,
                nullptr, 0, 1.f, (int)BS, HID, D, 1, 1);
    launch_gemm(stream, Xb, D, 0, 0, Wvb, HID, 0, 0, 0, Vb, 1, HID, 0, 0,
                nullptr, 0, 1.f, (int)BS, HID, D, 1, 1);

    // 3. scores[b,h,e,s] = (A^T @ K_bh^T)/16  (B operand = K_bh as [N,K], transB)
    launch_gemm(stream, AbT, D, 0, 0,
                Kb, HID, (long long)S * HID, D, 1,
                attnF, 0, S, (long long)H * D * S, (long long)D * S,
                nullptr, 0, 1.f / 16.f, D, S, D, B * H, H);

    // 4. softmax rows -> d_out f32 + bf16 copy
    k_softmax_row2048<<<(unsigned)(B * H * D), 256, 0, stream>>>(attnF, attnB);

    // 5. VV = attn @ V_bh
    launch_gemm(stream, attnB, S, (long long)H * D * S, (long long)D * S,
                Vb, HID, (long long)S * HID, D, 0,
                VVb, 1, D, (long long)H * D * D, (long long)D * D,
                nullptr, 0, 1.f, D, D, S, B * H, H);

    // 6. QA = Q_bh @ A  (writes attnB region)
    launch_gemm(stream, Qb, HID, (long long)S * HID, D,
                Ab, D, 0, 0, 0,
                QAb, 1, D, (long long)H * S * D, (long long)S * D,
                nullptr, 0, 1.f, S, D, D, B * H, H);

    // 7. context = QA @ VV^T -> merged heads (writes Kb region)
    launch_gemm(stream, QAb, D, (long long)H * S * D, (long long)S * D,
                VVb, D, (long long)H * D * D, (long long)D * D, 1,
                ctxB, 1, HID, (long long)S * HID, D,
                nullptr, 0, 1.f, S, D, D, B * H, H);

    // 8. ctx @ Wo -> f32
    launch_gemm(stream, ctxB, HID, 0, 0, Wob, D, 0, 0, 0,
                tmpF, 0, D, 0, 0, nullptr, 0, 1.f, (int)BS, D, HID, 1, 1);

    // 9. dual LayerNorm -> y2 (f32 + bf16)
    k_ln_dual<<<(unsigned)BS, 256, 0, stream>>>(tmpF, X, gAttn, bAttn, gEnc, bEnc, y2F, y2b);

    // 10. FFN1: relu(y2 @ W1 + b1) (writes Vb region)
    launch_gemm(stream, y2b, D, 0, 0, W1b, HID, 0, 0, 0,
                hB, 1, HID, 0, 0, b1, 1, 1.f, (int)BS, HID, D, 1, 1);

    // 11. FFN2: hidden @ W2 + b2 -> f32
    launch_gemm(stream, hB, HID, 0, 0, W2b, D, 0, 0, 0,
                ffnF, 0, D, 0, 0, b2, 0, 1.f, (int)BS, D, HID, 1, 1);

    // 12. final LayerNorm -> d_out
    k_ln_final<<<(unsigned)BS, 256, 0, stream>>>(ffnF, y2F, gFfn, bFfn, outP);
}
